// FFSPModel_22608707846856
// MI455X (gfx1250) — matrix operand order repaired and checked by reading, not compiled, not run on hardware
//
#include <hip/hip_runtime.h>
#include <hip/hip_bf16.h>

typedef __attribute__((ext_vector_type(16))) _Float16 v16h;
typedef __attribute__((ext_vector_type(8)))  _Float16 v8h;
typedef __attribute__((ext_vector_type(8)))  float    v8f;

#define B_  128
#define J_  100
#define M_  16
#define P_  128
#define D_  256
#define H_  16
#define QD_ 16
#define FF_ 512
#define L_  3

// ---------------------------------------------------------------------------
// f32 (K x N, possibly stacked) -> f16 transposed (N x K) per matrix
// grid = (ceil(K*N/256), nmat)
// ---------------------------------------------------------------------------
__global__ void f32_to_f16_t(const float* __restrict__ src, _Float16* __restrict__ dst,
                             int K, int N) {
  int mat = blockIdx.y;
  src += (size_t)mat * K * N;
  dst += (size_t)mat * K * N;
  int i = blockIdx.x * blockDim.x + threadIdx.x;
  if (i < K * N) {
    int k = i / N, nn = i - k * N;
    dst[(size_t)nn * K + k] = (_Float16)src[i];
  }
}

// ---------------------------------------------------------------------------
// Batched GEMM: C[b] = act(A[b](n x K) * W(K x N) + bias + res[b])
// Wt is the f16 TRANSPOSED weight (N x K). 4 waves/block, 16x64 tile/block.
// A panel (16 x K) staged in LDS as f16 (stride K+8 to avoid bank conflicts).
// Inner loop: 2x ds_load_b128 + 2x global_load_b128 + v_wmma_f32_16x16x32_f16.
// ---------------------------------------------------------------------------
__global__ void gemm_f16w(const float* __restrict__ A, const _Float16* __restrict__ Wt,
                          const float* __restrict__ bias, const float* __restrict__ res,
                          float* __restrict__ C, int n, int K, int N, int relu) {
  __shared__ _Float16 As[16 * (512 + 8)];
  const int pad = 8;
  int tid  = threadIdx.x;
  int lane = tid & 31;
  int wv   = tid >> 5;
  int tm = blockIdx.x * 16;
  int tn = blockIdx.y * 64 + wv * 16;
  int b  = blockIdx.z;
  const float* Ab = A + (size_t)b * n * K;
  float*       Cb = C + (size_t)b * n * N;
  const float* Rb = res ? res + (size_t)b * n * N : (const float*)0;

  int m  = lane & 15;
  int hi = lane >> 4;
  const _Float16* bw = Wt + (size_t)(tn + m) * K;
  __builtin_prefetch(bw, 0, 1);  // global_prefetch_b8: warm B panel during A staging

  // cooperative stage of A panel into LDS (f32 -> f16, float4 granularity)
  int quads = K >> 2;
  for (int idx = tid; idx < 16 * quads; idx += 128) {
    int row = idx / quads;
    int c4  = idx - row * quads;
    int rg  = tm + row; if (rg >= n) rg = n - 1;   // clamp: feeds only masked rows
    float4 f = *(const float4*)(Ab + (size_t)rg * K + (size_t)c4 * 4);
    union { _Float16 h[4]; uint2 u; } pk;
    pk.h[0] = (_Float16)f.x; pk.h[1] = (_Float16)f.y;
    pk.h[2] = (_Float16)f.z; pk.h[3] = (_Float16)f.w;
    *(uint2*)&As[row * (K + pad) + c4 * 4] = pk.u;
  }
  __syncthreads();

  const _Float16* ar = &As[m * (K + pad)];
  v8f acc = {};
#pragma unroll 4
  for (int k0 = 0; k0 < K; k0 += 32) {
    v8h a0 = *(const v8h*)&ar[k0 + hi * 8];        // K 0..7   / 8..15
    v8h a1 = *(const v8h*)&ar[k0 + 16 + hi * 8];   // K 16..23 / 24..31
    v8h b0 = *(const v8h*)&bw[k0 + hi * 8];        // K 0..7   / 8..15
    v8h b1 = *(const v8h*)&bw[k0 + 16 + hi * 8];
    v16h av = __builtin_shufflevector(a0, a1, 0,1,2,3,4,5,6,7,8,9,10,11,12,13,14,15);
    v16h bv = __builtin_shufflevector(b0, b1, 0,1,2,3,4,5,6,7,8,9,10,11,12,13,14,15);
    acc = __builtin_amdgcn_wmma_f32_16x16x32_f16(false, av, false, bv,
                                                 (short)0, acc, false, false);
  }

  int col = tn + m;
  float bi = bias ? bias[col] : 0.f;
#pragma unroll
  for (int r = 0; r < 8; ++r) {
    int row = tm + hi * 8 + r;
    if (row < n) {
      float v = acc[r] + bi;
      if (Rb) v += Rb[(size_t)row * N + col];
      if (relu) v = v > 0.f ? v : 0.f;
      Cb[(size_t)row * N + col] = v;
    }
  }
}

// ---------------------------------------------------------------------------
// Batched C[b] = scale * A[b](n x K) * Bm[b](nB x K)^T   (both f32 -> f16 tiles)
// ---------------------------------------------------------------------------
__global__ void gemm_bt(const float* __restrict__ A, const float* __restrict__ Bm,
                        float* __restrict__ C, int n, int nB, int K, float scale) {
  int lane = threadIdx.x;
  int tm = blockIdx.x * 16;
  int tn = blockIdx.y * 16;
  int b  = blockIdx.z;
  const float* Ab = A  + (size_t)b * n  * K;
  const float* Bb = Bm + (size_t)b * nB * K;
  float*       Cb = C  + (size_t)b * n  * nB;

  int m  = lane & 15;
  int hi = lane >> 4;
  int rA = tm + m; if (rA >= n)  rA = n  - 1;
  int cB = tn + m; if (cB >= nB) cB = nB - 1;
  const float* arow = Ab + (size_t)rA * K;
  const float* brow = Bb + (size_t)cB * K;

  v8f acc = {};
  for (int k0 = 0; k0 < K; k0 += 32) {
    v16h av, bv;
#pragma unroll
    for (int i = 0; i < 8; ++i) {
      av[i]     = (_Float16)arow[k0 + hi * 8 + i];
      av[8 + i] = (_Float16)arow[k0 + 16 + hi * 8 + i];
    }
#pragma unroll
    for (int i = 0; i < 8; ++i) {
      bv[i]     = (_Float16)brow[k0 + hi * 8 + i];
      bv[8 + i] = (_Float16)brow[k0 + 16 + hi * 8 + i];
    }
    acc = __builtin_amdgcn_wmma_f32_16x16x32_f16(false, av, false, bv,
                                                 (short)0, acc, false, false);
  }
  int col = tn + m;
#pragma unroll
  for (int r = 0; r < 8; ++r) {
    int row = tm + hi * 8 + r;
    if (row < n && col < nB) Cb[(size_t)row * nB + col] = acc[r] * scale;
  }
}

// ---------------------------------------------------------------------------
// Encoder mixed-score attention. grid=(nr, B), block = H*32 (one wave per head).
// ---------------------------------------------------------------------------
__global__ void enc_attn(const float* __restrict__ Q, const float* __restrict__ Kx,
                         const float* __restrict__ V, const float* __restrict__ cost,
                         int costRS, int costCS,
                         const float* __restrict__ m1w, const float* __restrict__ m1b,
                         const float* __restrict__ m2w, const float* __restrict__ m2b,
                         float* __restrict__ O, int nr, int nc) {
  __shared__ float wsh[H_][128];
  int lane = threadIdx.x & 31;
  int h    = threadIdx.x >> 5;
  int r    = blockIdx.x;
  int b    = blockIdx.y;

  const float* qp = Q + ((size_t)(b * nr + r)) * D_ + h * QD_;
  float qv[QD_];
#pragma unroll
  for (int d = 0; d < QD_; ++d) qv[d] = qp[d];

  float w1a[16], w1b[16], b1[16], w2[16];
#pragma unroll
  for (int mm = 0; mm < 16; ++mm) {
    w1a[mm] = m1w[(h * 2 + 0) * 16 + mm];
    w1b[mm] = m1w[(h * 2 + 1) * 16 + mm];
    b1[mm]  = m1b[h * 16 + mm];
    w2[mm]  = m2w[h * 16 + mm];
  }
  float b2 = m2b[h];
  const float* cb = cost + (size_t)b * J_ * M_ + (size_t)r * costRS;

  float sv[4];
  float mx = -3.0e38f;
#pragma unroll
  for (int it = 0; it < 4; ++it) {
    int c = lane + it * 32;
    float s = -3.0e38f;
    if (c < nc) {
      const float* kp = Kx + ((size_t)(b * nc + c)) * D_ + h * QD_;
      float dot = 0.f;
#pragma unroll
      for (int d = 0; d < QD_; ++d) dot += qv[d] * kp[d];
      dot *= 0.25f;
      float cv = cb[(size_t)c * costCS];
      float sc = b2;
#pragma unroll
      for (int mm = 0; mm < 16; ++mm) {
        float t = dot * w1a[mm] + cv * w1b[mm] + b1[mm];
        t = t > 0.f ? t : 0.f;
        sc += t * w2[mm];
      }
      s = sc;
    }
    sv[it] = s;
    mx = fmaxf(mx, s);
  }
  for (int o = 16; o; o >>= 1) mx = fmaxf(mx, __shfl_xor(mx, o, 32));
  float ev[4], sum = 0.f;
#pragma unroll
  for (int it = 0; it < 4; ++it) {
    int c = lane + it * 32;
    float e = (c < nc) ? __expf(sv[it] - mx) : 0.f;
    ev[it] = e; sum += e;
  }
  for (int o = 16; o; o >>= 1) sum += __shfl_xor(sum, o, 32);
  float inv = 1.f / sum;
#pragma unroll
  for (int it = 0; it < 4; ++it) {
    int c = lane + it * 32;
    if (c < nc) wsh[h][c] = ev[it] * inv;
  }
  __syncthreads();
  if (lane < QD_) {
    float acc = 0.f;
    for (int c = 0; c < nc; ++c)
      acc += wsh[h][c] * V[((size_t)(b * nc + c)) * D_ + h * QD_ + lane];
    O[((size_t)(b * nr + r)) * D_ + h * QD_ + lane] = acc;
  }
}

// ---------------------------------------------------------------------------
// Decoder attention. grid=(np, B), block=H*32. sc = q.k/4 + mask[b,p,j]
// ---------------------------------------------------------------------------
__global__ void dec_attn(const float* __restrict__ Q, const float* __restrict__ Kx,
                         const float* __restrict__ V, const float* __restrict__ mask,
                         float* __restrict__ O, int np, int nj) {
  __shared__ float wsh[H_][128];
  int lane = threadIdx.x & 31;
  int h    = threadIdx.x >> 5;
  int p    = blockIdx.x;
  int b    = blockIdx.y;

  const float* qp = Q + ((size_t)(b * np + p)) * D_ + h * QD_;
  float qv[QD_];
#pragma unroll
  for (int d = 0; d < QD_; ++d) qv[d] = qp[d];
  const float* mp = mask + ((size_t)(b * np + p)) * nj;

  float sv[4];
  float mx = -3.0e38f;
#pragma unroll
  for (int it = 0; it < 4; ++it) {
    int j = lane + it * 32;
    float s = -3.0e38f;
    if (j < nj) {
      const float* kp = Kx + ((size_t)(b * nj + j)) * D_ + h * QD_;
      float dot = 0.f;
#pragma unroll
      for (int d = 0; d < QD_; ++d) dot += qv[d] * kp[d];
      s = dot * 0.25f + mp[j];
    }
    sv[it] = s;
    mx = fmaxf(mx, s);
  }
  for (int o = 16; o; o >>= 1) mx = fmaxf(mx, __shfl_xor(mx, o, 32));
  float ev[4], sum = 0.f;
#pragma unroll
  for (int it = 0; it < 4; ++it) {
    int j = lane + it * 32;
    float e = (j < nj) ? __expf(sv[it] - mx) : 0.f;
    ev[it] = e; sum += e;
  }
  for (int o = 16; o; o >>= 1) sum += __shfl_xor(sum, o, 32);
  float inv = 1.f / sum;
#pragma unroll
  for (int it = 0; it < 4; ++it) {
    int j = lane + it * 32;
    if (j < nj) wsh[h][j] = ev[it] * inv;
  }
  __syncthreads();
  if (lane < QD_) {
    float acc = 0.f;
    for (int j = 0; j < nj; ++j)
      acc += wsh[h][j] * V[((size_t)(b * nj + j)) * D_ + h * QD_ + lane];
    O[((size_t)(b * np + p)) * D_ + h * QD_ + lane] = acc;
  }
}

// ---------------------------------------------------------------------------
// InstanceNorm1d over token dim. grid=(B), block=D. population variance.
// ---------------------------------------------------------------------------
__global__ void inorm(const float* __restrict__ X, const float* __restrict__ sc,
                      const float* __restrict__ bi, float* __restrict__ Y, int n) {
  int b = blockIdx.x;
  int d = threadIdx.x;
  const float* xb = X + (size_t)b * n * D_ + d;
  float s = 0.f, s2 = 0.f;
  for (int i = 0; i < n; ++i) {
    float v = xb[(size_t)i * D_];
    s += v; s2 += v * v;
  }
  float m   = s / (float)n;
  float var = s2 / (float)n - m * m;
  float inv = rsqrtf(var + 1e-5f);
  float g = sc[d], be = bi[d];
  float* yb = Y + (size_t)b * n * D_ + d;
  for (int i = 0; i < n; ++i)
    yb[(size_t)i * D_] = (xb[(size_t)i * D_] - m) * inv * g + be;
}

// jobs1 = concat(row, no_job). grid=(B), block=D
__global__ void build_jobs(const float* __restrict__ row, const float* __restrict__ nojob,
                           float* __restrict__ jobs) {
  int b = blockIdx.x, d = threadIdx.x;
  for (int i = 0; i < J_; ++i)
    jobs[((size_t)(b * (J_ + 1) + i)) * D_ + d] = row[((size_t)(b * J_ + i)) * D_ + d];
  jobs[((size_t)(b * (J_ + 1) + J_)) * D_ + d] = nojob[d];
}

// enc_mach gather. grid=(P,B), block=D
__global__ void gather_mach(const float* __restrict__ col, const int* __restrict__ idx,
                            float* __restrict__ Xq) {
  int p = blockIdx.x, b = blockIdx.y, d = threadIdx.x;
  int mi = idx[b * P_ + p];
  Xq[((size_t)(b * P_ + p)) * D_ + d] = col[((size_t)(b * M_ + mi)) * D_ + d];
}

// out = softmax(10*tanh(score) + mask) over nj. grid=(P,B), block=32 (one wave)
__global__ void final_softmax(const float* __restrict__ S, const float* __restrict__ mask,
                              float* __restrict__ out, int nj) {
  int lane = threadIdx.x;
  int p = blockIdx.x, b = blockIdx.y;
  const float* sp = S    + ((size_t)(b * P_ + p)) * nj;
  const float* mp = mask + ((size_t)(b * P_ + p)) * nj;
  float v[4], mx = -3.0e38f;
#pragma unroll
  for (int it = 0; it < 4; ++it) {
    int j = lane + it * 32;
    float x = -3.0e38f;
    if (j < nj) x = 10.f * tanhf(sp[j]) + mp[j];
    v[it] = x; mx = fmaxf(mx, x);
  }
  for (int o = 16; o; o >>= 1) mx = fmaxf(mx, __shfl_xor(mx, o, 32));
  float e[4], sum = 0.f;
#pragma unroll
  for (int it = 0; it < 4; ++it) {
    int j = lane + it * 32;
    e[it] = (j < nj) ? __expf(v[it] - mx) : 0.f;
    sum += e[it];
  }
  for (int o = 16; o; o >>= 1) sum += __shfl_xor(sum, o, 32);
  float inv = 1.f / sum;
  float* op = out + ((size_t)(b * P_ + p)) * nj;
#pragma unroll
  for (int it = 0; it < 4; ++it) {
    int j = lane + it * 32;
    if (j < nj) op[j] = e[it] * inv;
  }
}

// ---------------------------------------------------------------------------
extern "C" void kernel_launch(void* const* d_in, const int* in_sizes, int n_in,
                              void* d_out, int out_size, void* d_ws, size_t ws_size,
                              hipStream_t stream) {
  (void)in_sizes; (void)n_in; (void)out_size; (void)ws_size;
  const float* row_emb    = (const float*)d_in[0];
  const float* col_emb    = (const float*)d_in[1];
  const float* cost_mat   = (const float*)d_in[2];
  const float* ninf_mask  = (const float*)d_in[3];
  const float* enc_Wq     = (const float*)d_in[4];
  const float* enc_Wk     = (const float*)d_in[5];
  const float* enc_Wv     = (const float*)d_in[6];
  const float* mix1_w     = (const float*)d_in[7];
  const float* mix1_b     = (const float*)d_in[8];
  const float* mix2_w     = (const float*)d_in[9];
  const float* mix2_b     = (const float*)d_in[10];
  const float* comb_w     = (const float*)d_in[11];
  const float* comb_b     = (const float*)d_in[12];
  const float* norm1_s    = (const float*)d_in[13];
  const float* norm1_b    = (const float*)d_in[14];
  const float* ff_w1      = (const float*)d_in[15];
  const float* ff_b1      = (const float*)d_in[16];
  const float* ff_w2      = (const float*)d_in[17];
  const float* ff_b2      = (const float*)d_in[18];
  const float* norm2_s    = (const float*)d_in[19];
  const float* norm2_b    = (const float*)d_in[20];
  const float* no_job     = (const float*)d_in[21];
  const float* dec_Wq     = (const float*)d_in[22];
  const float* dec_Wk     = (const float*)d_in[23];
  const float* dec_Wv     = (const float*)d_in[24];
  const float* dec_comb_w = (const float*)d_in[25];
  const float* dec_comb_b = (const float*)d_in[26];
  const int*   machine_idx= (const int*)d_in[27];

  // workspace layout (bump allocator, 256B aligned)
  char* ws = (char*)d_ws;
  size_t off = 0;
  auto take = [&](size_t bytes) -> char* {
    char* p = ws + off;
    off += (bytes + 255) & ~(size_t)255;
    return p;
  };
  _Float16* wq16    = (_Float16*)take((size_t)L_*2*D_*D_*2);   // transposed (N,K)
  _Float16* wk16    = (_Float16*)take((size_t)L_*2*D_*D_*2);
  _Float16* wv16    = (_Float16*)take((size_t)L_*2*D_*D_*2);
  _Float16* comb16  = (_Float16*)take((size_t)L_*2*D_*D_*2);
  _Float16* ffw1_16 = (_Float16*)take((size_t)L_*2*D_*FF_*2);
  _Float16* ffw2_16 = (_Float16*)take((size_t)L_*2*FF_*D_*2);
  _Float16* dq16    = (_Float16*)take((size_t)D_*D_*2);
  _Float16* dk16    = (_Float16*)take((size_t)D_*D_*2);
  _Float16* dv16    = (_Float16*)take((size_t)D_*D_*2);
  _Float16* dcomb16 = (_Float16*)take((size_t)D_*D_*2);

  float* rowA  = (float*)take((size_t)B_*J_*D_*4);
  float* rowB  = (float*)take((size_t)B_*J_*D_*4);
  float* colA  = (float*)take((size_t)B_*M_*D_*4);
  float* colB  = (float*)take((size_t)B_*M_*D_*4);
  float* Qb    = (float*)take((size_t)B_*P_*D_*4);   // max n = 128
  float* Kb    = (float*)take((size_t)B_*P_*D_*4);
  float* Vb    = (float*)take((size_t)B_*P_*D_*4);
  float* Ob    = (float*)take((size_t)B_*P_*D_*4);
  float* Pre   = (float*)take((size_t)B_*P_*D_*4);
  float* FT    = (float*)take((size_t)B_*J_*FF_*4);
  float* jobs1 = (float*)take((size_t)B_*(J_+1)*D_*4);
  float* Xq    = (float*)take((size_t)B_*P_*D_*4);
  float* Mh    = (float*)take((size_t)B_*P_*D_*4);
  float* Sc    = (float*)take((size_t)B_*P_*(J_+1)*4);

  auto cvtT = [&](const float* s, _Float16* d, int K, int N, int nmat) {
    f32_to_f16_t<<<dim3((K * N + 255) / 256, nmat), dim3(256), 0, stream>>>(s, d, K, N);
  };
  cvtT(enc_Wq, wq16,    D_, D_,  L_ * 2);
  cvtT(enc_Wk, wk16,    D_, D_,  L_ * 2);
  cvtT(enc_Wv, wv16,    D_, D_,  L_ * 2);
  cvtT(comb_w, comb16,  D_, D_,  L_ * 2);
  cvtT(ff_w1,  ffw1_16, D_, FF_, L_ * 2);
  cvtT(ff_w2,  ffw2_16, FF_, D_, L_ * 2);
  cvtT(dec_Wq, dq16,    D_, D_,  1);
  cvtT(dec_Wk, dk16,    D_, D_,  1);
  cvtT(dec_Wv, dv16,    D_, D_,  1);
  cvtT(dec_comb_w, dcomb16, D_, D_, 1);

  auto gemm = [&](const float* A, const _Float16* Wt, const float* bias, const float* res,
                  float* C, int n, int K, int N, int relu) {
    gemm_f16w<<<dim3((n + 15) / 16, N / 64, B_), dim3(128), 0, stream>>>(
        A, Wt, bias, res, C, n, K, N, relu);
  };

  // ---------------- encoder ----------------
  const float* curRow = row_emb;
  const float* curCol = col_emb;
  float* rbufs[2] = { rowA, rowB };
  float* cbufs[2] = { colA, colB };
  for (int l = 0; l < L_; ++l) {
    float* nRow = rbufs[(l + 1) & 1];
    float* nCol = cbufs[(l + 1) & 1];
    for (int s = 0; s < 2; ++s) {
      int ls = l * 2 + s;
      const float* xr = (s == 0) ? curRow : curCol;
      const float* xc = (s == 0) ? curCol : curRow;
      int nr = (s == 0) ? J_ : M_;
      int nc = (s == 0) ? M_ : J_;
      float* outX = (s == 0) ? nRow : nCol;
      int costRS = (s == 0) ? M_ : 1;
      int costCS = (s == 0) ? 1 : M_;

      gemm(xr, wq16 + (size_t)ls * D_ * D_, (const float*)0, (const float*)0, Qb, nr, D_, D_, 0);
      gemm(xc, wk16 + (size_t)ls * D_ * D_, (const float*)0, (const float*)0, Kb, nc, D_, D_, 0);
      gemm(xc, wv16 + (size_t)ls * D_ * D_, (const float*)0, (const float*)0, Vb, nc, D_, D_, 0);

      enc_attn<<<dim3(nr, B_), dim3(H_ * 32), 0, stream>>>(
          Qb, Kb, Vb, cost_mat, costRS, costCS,
          mix1_w + (size_t)ls * H_ * 2 * 16, mix1_b + (size_t)ls * H_ * 16,
          mix2_w + (size_t)ls * H_ * 16,     mix2_b + (size_t)ls * H_,
          Ob, nr, nc);

      gemm(Ob, comb16 + (size_t)ls * D_ * D_, comb_b + (size_t)ls * D_, xr, Pre, nr, D_, D_, 0);
      inorm<<<dim3(B_), dim3(D_), 0, stream>>>(Pre, norm1_s + (size_t)ls * D_,
                                               norm1_b + (size_t)ls * D_, Ob, nr);
      gemm(Ob, ffw1_16 + (size_t)ls * D_ * FF_, ff_b1 + (size_t)ls * FF_, (const float*)0,
           FT, nr, D_, FF_, 1);
      gemm(FT, ffw2_16 + (size_t)ls * FF_ * D_, ff_b2 + (size_t)ls * D_, Ob, Pre, nr, FF_, D_, 0);
      inorm<<<dim3(B_), dim3(D_), 0, stream>>>(Pre, norm2_s + (size_t)ls * D_,
                                               norm2_b + (size_t)ls * D_, outX, nr);
    }
    curRow = nRow;
    curCol = nCol;
  }

  // ---------------- decoder ----------------
  build_jobs<<<dim3(B_), dim3(D_), 0, stream>>>(curRow, no_job, jobs1);
  gemm(jobs1, dk16, (const float*)0, (const float*)0, Kb, J_ + 1, D_, D_, 0);
  gemm(jobs1, dv16, (const float*)0, (const float*)0, Vb, J_ + 1, D_, D_, 0);
  gather_mach<<<dim3(P_, B_), dim3(D_), 0, stream>>>(curCol, machine_idx, Xq);
  gemm(Xq, dq16, (const float*)0, (const float*)0, Qb, P_, D_, D_, 0);
  dec_attn<<<dim3(P_, B_), dim3(H_ * 32), 0, stream>>>(Qb, Kb, Vb, ninf_mask, Ob, P_, J_ + 1);
  gemm(Ob, dcomb16, dec_comb_b, (const float*)0, Mh, P_, D_, D_, 0);
  gemm_bt<<<dim3(P_ / 16, (J_ + 1 + 15) / 16, B_), dim3(32), 0, stream>>>(
      Mh, jobs1, Sc, P_, J_ + 1, D_, 1.f / 16.f);
  final_softmax<<<dim3(P_, B_), dim3(32), 0, stream>>>(Sc, ninf_mask, (float*)d_out, J_ + 1);
}
